// cross_att_blocked_47691316855429
// MI455X (gfx1250) — matrix operand order repaired and checked by reading, not compiled, not run on hardware
//
#include <hip/hip_runtime.h>
#include <hip/hip_bf16.h>
#include <math.h>

// ---- problem constants (from reference setup_inputs) ----
#define BB 64
#define VV 1024
#define TT 64
#define EE 512
#define HH 3
#define KC 520            // E + 8 coord features
#define KCPAD 544         // 17 * 32
#define EPSV 1e-8f
#define SMOOTHV 9.0f

typedef __attribute__((ext_vector_type(16))) __bf16 v16bf;
typedef __attribute__((ext_vector_type(8)))  __bf16 v8bf;
typedef __attribute__((ext_vector_type(8)))  float  v8f;

// ---------------- WMMA fragment helpers (ISA 7.12.2 layouts, wave32) ----------------
// A matrix 16x32 bf16: lane<16 -> M=lane, halves {0..7}=K 0..7, {8..15}=K 16..23
//                      lane>=16 -> M=lane-16, halves {0..7}=K 8..15, {8..15}=K 24..31
__device__ inline v16bf load_frag_A(const __bf16* lds, int ld, int m0, int lane) {
    int half = (lane >> 4) & 1;
    int r    = lane & 15;
    const __bf16* base = lds + (m0 + r) * ld;
    v8bf lo = *(const v8bf*)(base + half * 8);
    v8bf hi = *(const v8bf*)(base + 16 + half * 8);
    v16bf out;
#pragma unroll
    for (int i = 0; i < 8; i++) { out[i] = lo[i]; out[8 + i] = hi[i]; }
    return out;
}

// B matrix 32x16 bf16 stored in LDS as [N][K] (column-major K x N):
// lane<16 -> N=lane, halves {0..7}=K 0..7, {8..15}=K 16..23; lane>=16 -> N=lane-16, K 8..15, 24..31
__device__ inline v16bf load_frag_B(const __bf16* lds, int ld, int n0, int lane) {
    int half = (lane >> 4) & 1;
    int n    = lane & 15;
    const __bf16* base = lds + (n0 + n) * ld;
    v8bf lo = *(const v8bf*)(base + half * 8);
    v8bf hi = *(const v8bf*)(base + 16 + half * 8);
    v16bf out;
#pragma unroll
    for (int i = 0; i < 8; i++) { out[i] = lo[i]; out[8 + i] = hi[i]; }
    return out;
}
// C/D 16x16 f32: VGPR i: lanes<16 -> M=i, lanes>=16 -> M=8+i; N = lane&15

// ---------------- operand prep: concat(fvisu,coord) zero-padded -> bf16 ----------------
__global__ __launch_bounds__(256) void prep_map_kernel(const float* __restrict__ fvisu,
                                                       const float* __restrict__ coord,
                                                       __bf16* __restrict__ bmap) {
    long row = blockIdx.x;
    const float* fr = fvisu + row * (long)EE;
    const float* cr = coord + row * 8L;
    __bf16* o = bmap + row * (long)KCPAD;
    for (int k = threadIdx.x; k < KCPAD; k += 256) {
        float v = (k < EE) ? fr[k] : ((k < KC) ? cr[k - EE] : 0.f);
        o[k] = (__bf16)v;
    }
}

// W (H*E rows, 520 cols) -> bf16 zero-padded to 544
__global__ __launch_bounds__(256) void prep_w_kernel(const float* __restrict__ W,
                                                     __bf16* __restrict__ bw) {
    long row = blockIdx.x;
    const float* src = W + row * (long)KC;
    __bf16* o = bw + row * (long)KCPAD;
    for (int k = threadIdx.x; k < KCPAD; k += 256) {
        o[k] = (__bf16)((k < KC) ? src[k] : 0.f);
    }
}

// ---------------- l2norm over last axis (C == 512): optional f32 and bf16 outputs ----------------
__global__ __launch_bounds__(256) void l2norm_dual_kernel(const float* __restrict__ in,
                                                          float* __restrict__ outf,
                                                          __bf16* __restrict__ outb) {
    __shared__ float red[256];
    long row = blockIdx.x;
    const float* r = in + row * (long)EE;
    int t = threadIdx.x;
    float v0 = r[t], v1 = r[t + 256];
    float a0 = v0 + EPSV, a1 = v1 + EPSV;
    red[t] = a0 * a0 + a1 * a1;
    __syncthreads();
#pragma unroll
    for (int off = 128; off > 0; off >>= 1) {
        if (t < off) red[t] += red[t + off];
        __syncthreads();
    }
    float inv = 1.f / (sqrtf(red[0]) + EPSV);
    if (outf) { float* w = outf + row * (long)EE; w[t] = v0 * inv; w[t + 256] = v1 * inv; }
    if (outb) { __bf16* w = outb + row * (long)EE; w[t] = (__bf16)(v0 * inv); w[t + 256] = (__bf16)(v1 * inv); }
}

// ---------------- fused per-head projection + bias + relu + row l2norm -> bf16 ----------------
// block tile: 64(M) x 512(N, full); 8 waves, wave w owns N cols [w*64, w*64+64); grid.x = B*V/64
__global__ __launch_bounds__(256) void proj_fused_kernel(const __bf16* __restrict__ bmap,
                                                         const __bf16* __restrict__ bw,
                                                         const float* __restrict__ bn,
                                                         __bf16* __restrict__ bf_f) {
    __shared__ __align__(16) __bf16 sA[64 * 40];
    __shared__ __align__(16) __bf16 sB[512 * 40];
    __shared__ float rowsum[64];
    int t = threadIdx.x, lane = t & 31, wave = t >> 5;
    long m0 = (long)blockIdx.x * 64;
    v8f acc[4][4] = {};

    for (int kk = 0; kk < KCPAD; kk += 32) {
        __syncthreads();
        {   // A: 64 rows x 32 k, pure bf16 copy (one 16B chunk / thread)
            int row = t >> 2, seg = (t & 3) * 8;
            const __bf16* src = bmap + (m0 + row) * (long)KCPAD + kk + seg;
            *(v8bf*)(sA + row * 40 + seg) = *(const v8bf*)src;
            if (kk + 32 < KCPAD) __builtin_prefetch(src + 32, 0, 1);
        }
        {   // B: 512 rows x 32 k (8 x 16B chunks / thread)
            int row = t >> 2, seg = (t & 3) * 8;
#pragma unroll
            for (int i = 0; i < 8; i++) {
                int rr = row + i * 64;
                const __bf16* src = bw + (long)rr * KCPAD + kk + seg;
                *(v8bf*)(sB + rr * 40 + seg) = *(const v8bf*)src;
            }
            if (kk + 32 < KCPAD) __builtin_prefetch(bw + (long)row * KCPAD + kk + 32 + seg, 0, 1);
        }
        __syncthreads();
#pragma unroll
        for (int mi = 0; mi < 4; mi++) {
            v16bf a = load_frag_A(sA, 40, mi * 16, lane);
#pragma unroll
            for (int ni = 0; ni < 4; ni++) {
                v16bf b = load_frag_B(sB, 40, wave * 64 + ni * 16, lane);
                acc[mi][ni] = __builtin_amdgcn_wmma_f32_16x16x32_bf16(
                    false, a, false, b, (short)0, acc[mi][ni], false, false);
            }
        }
    }

    // epilogue: bias + relu, then fused row l2norm (sum over all 512 cols)
    int half = (lane >> 4) & 1, nc = lane & 15;
    if (t < 64) rowsum[t] = 0.f;
    __syncthreads();
#pragma unroll
    for (int mi = 0; mi < 4; mi++) {
#pragma unroll
        for (int r = 0; r < 8; r++) {
            float ssq = 0.f;
#pragma unroll
            for (int ni = 0; ni < 4; ni++) {
                float bias = bn[wave * 64 + ni * 16 + nc];
                float v = acc[mi][ni][r] + bias;
                v = v > 0.f ? v : 0.f;
                acc[mi][ni][r] = v;
                float e = v + EPSV;
                ssq += e * e;
            }
            // reduce across the 16 lanes sharing this row (xor masks stay inside the half)
#pragma unroll
            for (int m = 8; m > 0; m >>= 1) ssq += __shfl_xor(ssq, m, 32);
            if (nc == 0) atomicAdd(&rowsum[mi * 16 + half * 8 + r], ssq);
        }
    }
    __syncthreads();
#pragma unroll
    for (int mi = 0; mi < 4; mi++)
#pragma unroll
        for (int r = 0; r < 8; r++) {
            int rl = mi * 16 + half * 8 + r;
            float inv = 1.f / (sqrtf(rowsum[rl]) + EPSV);
#pragma unroll
            for (int ni = 0; ni < 4; ni++) {
                int col = wave * 64 + ni * 16 + nc;
                bf_f[(m0 + rl) * (long)EE + col] = (__bf16)(acc[mi][ni][r] * inv);
            }
        }
}

// ---------------- attn[b,s,q] = dot(ctx[b,s,:], qry[b,q,:]) over d=512 (all-bf16 operands) ----------------
// grid: (Q/64, S/64, B); block 256; wave tile 16(M) x 32(N)
__global__ __launch_bounds__(256) void attn_gemm_kernel(const __bf16* __restrict__ ctx,
                                                        const __bf16* __restrict__ qry,
                                                        float* __restrict__ out,
                                                        int S, int Q) {
    __shared__ __align__(16) __bf16 sA[64 * 40];
    __shared__ __align__(16) __bf16 sB[64 * 40];
    int t = threadIdx.x, lane = t & 31, wave = t >> 5;
    int wm = wave & 3, wn = wave >> 2;
    int b = blockIdx.z, s0 = blockIdx.y * 64, q0 = blockIdx.x * 64;
    const __bf16* cb = ctx + ((long)b * S + s0) * EE;
    const __bf16* qb = qry + ((long)b * Q + q0) * EE;
    v8f acc[2] = {};

    for (int kk = 0; kk < EE; kk += 32) {
        __syncthreads();
        int row = t >> 2, seg = (t & 3) * 8;
        const __bf16* sa = cb + (long)row * EE + kk + seg;
        const __bf16* sb = qb + (long)row * EE + kk + seg;
        *(v8bf*)(sA + row * 40 + seg) = *(const v8bf*)sa;
        *(v8bf*)(sB + row * 40 + seg) = *(const v8bf*)sb;
        if (kk + 32 < EE) { __builtin_prefetch(sa + 32, 0, 1); __builtin_prefetch(sb + 32, 0, 1); }
        __syncthreads();
        v16bf a = load_frag_A(sA, 40, wm * 16, lane);
#pragma unroll
        for (int j = 0; j < 2; j++) {
            v16bf bfrag = load_frag_B(sB, 40, wn * 32 + j * 16, lane);
            acc[j] = __builtin_amdgcn_wmma_f32_16x16x32_bf16(
                false, a, false, bfrag, (short)0, acc[j], false, false);
        }
    }
    int half = (lane >> 4) & 1, nc = lane & 15;
#pragma unroll
    for (int j = 0; j < 2; j++)
#pragma unroll
        for (int r = 0; r < 8; r++) {
            long row = (long)b * S + s0 + wm * 16 + half * 8 + r;
            out[row * (long)Q + q0 + wn * 32 + j * 16 + nc] = acc[j][r];
        }
}

// ---------------- softmax over Q per (b,s) row, then *= (1 - weight) ----------------
__global__ __launch_bounds__(256) void softmax_mask_kernel(float* __restrict__ attn,
                                                           const float* __restrict__ weight,
                                                           int Q) {
    __shared__ float red[256];
    long row = blockIdx.x;
    float* a = attn + row * (long)Q;
    const float* w = weight + row * (long)Q;
    int t = threadIdx.x;
    float vals[4];
    int cnt = 0;
    float m = -3.402823466e38f;
    for (int q = t; q < Q; q += 256) { float v = a[q]; vals[cnt++] = v; m = fmaxf(m, v); }
    red[t] = m; __syncthreads();
#pragma unroll
    for (int off = 128; off > 0; off >>= 1) { if (t < off) red[t] = fmaxf(red[t], red[t + off]); __syncthreads(); }
    m = red[0]; __syncthreads();
    float s = 0.f; cnt = 0;
    for (int q = t; q < Q; q += 256) { float e = __expf(vals[cnt] - m); vals[cnt] = e; s += e; cnt++; }
    red[t] = s; __syncthreads();
#pragma unroll
    for (int off = 128; off > 0; off >>= 1) { if (t < off) red[t] += red[t + off]; __syncthreads(); }
    s = red[0];
    cnt = 0;
    for (int q = t; q < Q; q += 256) { a[q] = (vals[cnt] / s) * (1.f - w[q]); cnt++; }
}

// ---- per (b,q): cos = max_s attn[b,s,q]; p[b,q,s] = softmax_s(9*attn); f32 + bf16 outputs ----
__global__ __launch_bounds__(256) void colmax_softmax_kernel(const float* __restrict__ attn,
                                                             float* __restrict__ p,
                                                             __bf16* __restrict__ pbf,
                                                             float* __restrict__ cosout,
                                                             int S, int Q, int total) {
    long col = (long)blockIdx.x * 256 + threadIdx.x;
    if (col >= total) return;
    int b = (int)(col / Q), q = (int)(col % Q);
    const float* a = attn + (long)b * S * Q + q;
    float m = -3.402823466e38f;
    for (int s = 0; s < S; s++) m = fmaxf(m, a[(long)s * Q]);
    cosout[col] = m;
    float sum = 0.f;
    for (int s = 0; s < S; s++) sum += __expf(SMOOTHV * (a[(long)s * Q] - m));
    float inv = 1.f / sum;
    float* pr = p + col * (long)S;
    __bf16* pb = pbf + col * (long)S;
    for (int s = 0; s < S; s++) {
        float v = __expf(SMOOTHV * (a[(long)s * Q] - m)) * inv;
        pr[s] = v;
        pb[s] = (__bf16)v;
    }
}

// ---------------- running elementwise max (flat == reference's raw-reshape view) ----------------
__global__ __launch_bounds__(256) void maxupdate_kernel(float* __restrict__ dst,
                                                        const float* __restrict__ src,
                                                        int n, int first) {
    long i = (long)blockIdx.x * 256 + threadIdx.x;
    if (i >= n) return;
    dst[i] = first ? src[i] : fmaxf(dst[i], src[i]);
}

// ---------------- out[b,q,d] (+)= sum_s p[b,q,s]*ctx[b,s,d]  (M=Q,K=S,N=E, bf16 operands) ----------------
// grid: (E/64, Q/64, B); block 256
__global__ __launch_bounds__(256) void weighted_gemm_kernel(const __bf16* __restrict__ p,
                                                            const __bf16* __restrict__ ctx,
                                                            float* __restrict__ out,
                                                            int Q, int S, int accumulate) {
    __shared__ __align__(16) __bf16 sA[64 * 40];
    __shared__ __align__(16) __bf16 sB[64 * 40];   // stored [n][k] (transposed)
    int t = threadIdx.x, lane = t & 31, wave = t >> 5;
    int wm = wave & 3, wn = wave >> 2;
    int b = blockIdx.z, q0 = blockIdx.y * 64, n0 = blockIdx.x * 64;
    const __bf16* pb = p + ((long)b * Q + q0) * S;
    const __bf16* cb = ctx + (long)b * S * EE;
    v8f acc[2] = {};

    for (int kk = 0; kk < S; kk += 32) {
        __syncthreads();
        {   // A: p rows (K=S contiguous), straight bf16 copy
            int row = t >> 2, seg = (t & 3) * 8;
            const __bf16* src = pb + (long)row * S + kk + seg;
            *(v8bf*)(sA + row * 40 + seg) = *(const v8bf*)src;
            if (kk + 32 < S) __builtin_prefetch(src + 32, 0, 1);
        }
        {   // B: ctx is [K][N] row-major -> transpose into LDS [N][K]
            int krow = t >> 3, seg = (t & 7) * 8;
            const __bf16* src = cb + (long)(kk + krow) * EE + n0 + seg;
            v8bf v = *(const v8bf*)src;
#pragma unroll
            for (int j = 0; j < 8; j++)
                sB[(seg + j) * 40 + krow] = v[j];
            if (kk + 32 < S) __builtin_prefetch(cb + (long)(kk + 32 + krow) * EE + n0 + seg, 0, 1);
        }
        __syncthreads();
        v16bf a = load_frag_A(sA, 40, wm * 16, lane);
#pragma unroll
        for (int j = 0; j < 2; j++) {
            v16bf bfrag = load_frag_B(sB, 40, wn * 32 + j * 16, lane);
            acc[j] = __builtin_amdgcn_wmma_f32_16x16x32_bf16(
                false, a, false, bfrag, (short)0, acc[j], false, false);
        }
    }
    int half = (lane >> 4) & 1, nc = lane & 15;
#pragma unroll
    for (int j = 0; j < 2; j++)
#pragma unroll
        for (int r = 0; r < 8; r++) {
            long row = (long)b * Q + q0 + wm * 16 + half * 8 + r;
            long idx = row * (long)EE + n0 + wn * 32 + j * 16 + nc;
            out[idx] = accumulate ? (out[idx] + acc[j][r]) : acc[j][r];
        }
}

// ---------------- host orchestration ----------------
extern "C" void kernel_launch(void* const* d_in, const int* in_sizes, int n_in,
                              void* d_out, int out_size, void* d_ws, size_t ws_size,
                              hipStream_t stream) {
    const float* fvisu     = (const float*)d_in[0];
    const float* raw_fword = (const float*)d_in[1];
    const float* coord     = (const float*)d_in[2];
    const float* aw_txt    = (const float*)d_in[3];
    const float* aw_visu   = (const float*)d_in[4];
    /* d_in[5] word_mask unused by reference */
    const float* W         = (const float*)d_in[6];
    const float* bvec      = (const float*)d_in[7];

    float* out = (float*)d_out;

    // ---- workspace layout ----
    char* wsb = (char*)d_ws;
    float* ws_fword_n = (float*)wsb;                 wsb += sizeof(float) * (long)BB * TT * EE;
    float* ws_attn    = (float*)wsb;                 wsb += sizeof(float) * (long)BB * TT * VV;
    float* ws_p       = (float*)wsb;                 wsb += sizeof(float) * (long)BB * VV * TT;
    __bf16* bf_map     = (__bf16*)wsb;               wsb += sizeof(__bf16) * (long)BB * VV * KCPAD;
    __bf16* bf_W       = (__bf16*)wsb;               wsb += sizeof(__bf16) * (long)HH * EE * KCPAD;
    __bf16* bf_fword_n = (__bf16*)wsb;               wsb += sizeof(__bf16) * (long)BB * TT * EE;
    __bf16* bf_ftxt    = (__bf16*)wsb;               wsb += sizeof(__bf16) * (long)BB * TT * EE;
    __bf16* bf_fvisu_n = (__bf16*)wsb;               wsb += sizeof(__bf16) * (long)BB * VV * EE;
    __bf16* bf_f       = (__bf16*)wsb;               wsb += sizeof(__bf16) * (long)BB * VV * EE;
    __bf16* bf_p       = (__bf16*)wsb;               wsb += sizeof(__bf16) * (long)BB * VV * TT;

    // ---- output layout (floats, return order) ----
    float* o_att_visu  = out;                                   // B*V*E
    float* o_att_txt   = o_att_visu + (long)BB * VV * EE;       // B*T*E
    float* o_cos_t     = o_att_txt  + (long)BB * TT * EE;       // 1*B*T
    float* o_cos_visu  = o_cos_t    + (long)BB * TT;            // H*B*V
    float* o_aw_t      = o_cos_visu + (long)HH * BB * VV;       // B*V*T
    float* o_visu_coll = o_aw_t     + (long)BB * VV * TT;       // B*T*V

    // ---- operand prep (bf16 once, reused by all heads) ----
    prep_map_kernel<<<BB * VV, 256, 0, stream>>>(fvisu, coord, bf_map);
    prep_w_kernel<<<HH * EE, 256, 0, stream>>>(W, bf_W);
    l2norm_dual_kernel<<<BB * TT, 256, 0, stream>>>(raw_fword, ws_fword_n, bf_fword_n);
    l2norm_dual_kernel<<<BB * VV, 256, 0, stream>>>(fvisu, nullptr, bf_fvisu_n);
    l2norm_dual_kernel<<<BB * TT, 256, 0, stream>>>(ws_fword_n, nullptr, bf_ftxt); // double-norm, as in torch

    // ---- visual branch: 3 heads ----
    for (int h = 0; h < HH; h++) {
        proj_fused_kernel<<<(BB * VV) / 64, 256, 0, stream>>>(
            bf_map, bf_W + (long)h * EE * KCPAD, bvec + (long)h * EE, bf_f);

        dim3 ga(VV / 64, TT / 64, BB);
        attn_gemm_kernel<<<ga, 256, 0, stream>>>(bf_fword_n, bf_f, ws_attn, TT, VV);

        softmax_mask_kernel<<<BB * TT, 256, 0, stream>>>(
            ws_attn, (h == 0) ? aw_visu : o_visu_coll, VV);

        colmax_softmax_kernel<<<(BB * VV) / 256, 256, 0, stream>>>(
            ws_attn, ws_p, bf_p, o_cos_visu + (long)h * BB * VV, TT, VV, BB * VV);

        maxupdate_kernel<<<(BB * TT * VV) / 256, 256, 0, stream>>>(
            o_visu_coll, ws_p, BB * TT * VV, h == 0);

        dim3 gw(EE / 64, VV / 64, BB);
        weighted_gemm_kernel<<<gw, 256, 0, stream>>>(bf_p, bf_fword_n, o_att_visu,
                                                     VV, TT, h > 0);
    }

    // ---- text branch ----
    dim3 gat(TT / 64, VV / 64, BB);
    attn_gemm_kernel<<<gat, 256, 0, stream>>>(bf_fvisu_n, bf_ftxt, ws_attn, VV, TT);
    softmax_mask_kernel<<<BB * VV, 256, 0, stream>>>(ws_attn, aw_txt, TT);
    colmax_softmax_kernel<<<(BB * TT) / 256, 256, 0, stream>>>(
        ws_attn, o_aw_t, bf_p, o_cos_t, VV, TT, BB * TT);
    dim3 gwt(EE / 64, TT / 64, BB);
    weighted_gemm_kernel<<<gwt, 256, 0, stream>>>(bf_p, bf_fvisu_n, o_att_txt,
                                                  TT, VV, 0);

    // final in-place l2norms (f32 outputs)
    l2norm_dual_kernel<<<BB * VV, 256, 0, stream>>>(o_att_visu, o_att_visu, nullptr);
    l2norm_dual_kernel<<<BB * TT, 256, 0, stream>>>(o_att_txt, o_att_txt, nullptr);
}
